// Policy_LSTM_90477781058031
// MI455X (gfx1250) — compile-verified, matrix operand order checked by reading, not run on hardware
//
#include <hip/hip_runtime.h>
#include <hip/hip_bf16.h>

// ---------------------------------------------------------------------------
// Policy LSTM for MI455X (gfx1250, wave32, WMMA).
// Batch-parallel persistent LSTM: each workgroup owns 16 batch rows and loops
// over all T=1024 timesteps privately (recurrence is independent per batch
// row -> no inter-WG sync, one kernel launch). Gate matmuls use
// v_wmma_f32_16x16x32_bf16 with fp32 accumulation; weights are pre-swizzled
// into the exact WMMA B-operand lane layout (two coalesced b128 loads per
// fragment from L2). 512 threads / 16 waves per WG, bounded unrolling to keep
// VGPR pressure low and avoid scratch spills.
// ---------------------------------------------------------------------------

typedef __bf16 bf16_t;
typedef __attribute__((ext_vector_type(16))) __bf16 v16bf;
typedef __attribute__((ext_vector_type(8)))  __bf16 bf16x8;
typedef __attribute__((ext_vector_type(8)))  float  v8f;

#define HSZ   256          // hidden size
#define FH    1024         // 4*H (gate width)
#define FIN   10           // input features
#define BSZ   512          // batch
#define TSZ   1024         // timesteps
#define BT    16           // batch rows per workgroup
#define NWG   (BSZ / BT)   // 32 workgroups
#define BLK   512          // threads per workgroup (16 waves)

union Frag { v16bf v; bf16x8 h[2]; };

// A operand (16x32 bf16, M=batch rows, K=hidden): row = lane&15,
// lane>=16 shifts K by +8; halves 0..7 -> K=k0+khi+0..7, halves 8..15 ->
// K=k0+16+khi+0..7. Both runs are 8 contiguous bf16 -> two 16B LDS loads.
__device__ inline v16bf load_A_lds(const bf16_t* hbuf, int lane, int k0) {
  int row = lane & 15;
  int khi = (lane >> 4) << 3;
  const bf16_t* p = hbuf + row * HSZ + k0 + khi;
  Frag a;
  a.h[0] = *(const bf16x8*)(p);
  a.h[1] = *(const bf16x8*)(p + 16);
  return a.v;
}

// B operand from pre-swizzled weights: each 512-element (1KB) block stores
// lane l's 16 halves contiguously at block*512 + l*16.
__device__ inline v16bf load_B_pk(const bf16_t* pk, int block, int lane) {
  const bf16_t* p = pk + block * 512 + lane * 16;
  Frag b;
  b.h[0] = *(const bf16x8*)(p);
  b.h[1] = *(const bf16x8*)(p + 8);
  return b.v;
}

__device__ inline v8f wmma_bf16(v16bf a, v16bf b, v8f c) {
  return __builtin_amdgcn_wmma_f32_16x16x32_bf16(false, a, false, b,
                                                 (short)0, c, false, false);
}

__device__ inline float sigm(float x) { return 1.0f / (1.0f + __expf(-x)); }
__device__ inline float tanh_fast(float x) {
  float ax = fabsf(x);
  float e  = __expf(-2.0f * ax);           // in (0,1], never overflows
  float t  = (1.0f - e) / (1.0f + e);
  return copysignf(t, x);
}

// ---------------------------------------------------------------------------
// Prep: swizzle fp32 W[rows][K] (row-major, rows = output/gate index) into the
// WMMA B-operand bf16 layout. Block = (col_tile, k_block); element index
// e = block*512 + lane*16 + half; k = kb*32 + (h&7) + 16*(h>>3) + 8*(lane>>4);
// col = ct*16 + (lane&15).  B[k][n] == W[n][k].
// ---------------------------------------------------------------------------
__global__ void pack_wmma_b(const float* __restrict__ src, bf16_t* __restrict__ dst,
                            int K, int nkb, int total) {
  for (int e = blockIdx.x * blockDim.x + threadIdx.x; e < total;
       e += gridDim.x * blockDim.x) {
    int block  = e >> 9;
    int within = e & 511;
    int l  = within >> 4;
    int h  = within & 15;
    int ct = block / nkb;
    int kb = block - ct * nkb;
    int col = ct * 16 + (l & 15);
    int k   = kb * 32 + (h & 7) + ((h >> 3) << 4) + ((l >> 4) << 3);
    dst[e] = (bf16_t)src[col * K + k];
  }
}

__global__ void prep_bias(const float* __restrict__ bi0, const float* __restrict__ bh0,
                          const float* __restrict__ bi1, const float* __restrict__ bh1,
                          float* __restrict__ bias0, float* __restrict__ bias1) {
  int i = blockIdx.x * blockDim.x + threadIdx.x;
  if (i < FH) { bias0[i] = bi0[i] + bh0[i]; bias1[i] = bi1[i] + bh1[i]; }
}

// ---------------------------------------------------------------------------
// Persistent 2-layer LSTM + MLP head. 32 blocks x 512 threads (16 waves).
// ---------------------------------------------------------------------------
__global__ __launch_bounds__(BLK, 1)
void lstm_persistent(const float*  __restrict__ x,        // [B,T,FIN]
                     const float*  __restrict__ w_ih0,    // [FH,FIN] fp32
                     const bf16_t* __restrict__ pk_whh0,  // swizzled [H x FH]
                     const bf16_t* __restrict__ pk_wih1,
                     const bf16_t* __restrict__ pk_whh1,
                     const bf16_t* __restrict__ pk_adv1,  // swizzled [H x H]
                     const float*  __restrict__ bias0,    // [FH]
                     const float*  __restrict__ bias1,    // [FH]
                     const float*  __restrict__ adv_b1,   // [H]
                     const float*  __restrict__ adv_w2,   // [H]
                     const float*  __restrict__ adv_b2,   // [1]
                     float*        __restrict__ scores)   // [T*B] time-major
{
  __shared__ float  gates[BT * FH];        // 64KB: gate matmul output / hidden
  __shared__ bf16_t h1buf[BT * HSZ];       // 8KB
  __shared__ bf16_t h2buf[BT * HSZ];       // 8KB
  __shared__ float  wih0_s[FH * FIN];      // 40KB (K=10 input matmul, VALU)
  __shared__ float  bias0_s[FH];
  __shared__ float  bias1_s[FH];
  __shared__ float  advb1_s[HSZ];
  __shared__ float  advw2_s[HSZ];
  __shared__ float  xt[BT][FIN];
  __shared__ float  sp[BLK];

  const int tid  = threadIdx.x;
  const int lane = tid & 31;
  const int wave = tid >> 5;               // 0..15
  const int row0 = blockIdx.x * BT;

  for (int i = tid; i < FH * FIN; i += BLK) wih0_s[i] = w_ih0[i];
  for (int i = tid; i < FH; i += BLK) { bias0_s[i] = bias0[i]; bias1_s[i] = bias1[i]; }
  for (int i = tid; i < HSZ; i += BLK) { advb1_s[i] = adv_b1[i]; advw2_s[i] = adv_w2[i]; }
  for (int i = tid; i < BT * HSZ; i += BLK) { h1buf[i] = (bf16_t)0.0f; h2buf[i] = (bf16_t)0.0f; }

  // Cell state lives in registers: thread owns row eb, 8 consecutive columns.
  float c1r[8], c2r[8];
#pragma unroll
  for (int r = 0; r < 8; ++r) { c1r[r] = 0.0f; c2r[r] = 0.0f; }
  const int eb  = tid >> 5;                // 0..15 batch row in tile
  const int ej0 = (tid & 31) * 8;          // 8 hidden columns per thread

  const int colhalf = lane & 15;           // N within a tile (C/D lane layout)
  const int mbase   = (lane >> 4) << 3;    // M offset for this lane's 8 floats
  const float b2    = adv_b2[0];

  __syncthreads();

  for (int t = 0; t < TSZ; ++t) {
    // ---- stage x_t tile ----
    if (tid < BT * FIN) {
      int b = tid / FIN, k = tid - b * FIN;
      xt[b][k] = x[((size_t)(row0 + b) * TSZ + t) * FIN + k];
    }
    __syncthreads();

    // ---- layer0: gates = x@Wih0^T (VALU) + h1@Whh0^T (WMMA) + bias0 ----
#pragma unroll 1
    for (int tp = 0; tp < 2; ++tp) {
      int ct0 = wave * 4 + tp * 2, ct1 = ct0 + 1;
      int cA = ct0 * 16 + colhalf, cB = ct1 * 16 + colhalf;
      v8f acc0, acc1;
      float bA = bias0_s[cA], bB = bias0_s[cB];
#pragma unroll
      for (int r = 0; r < 8; ++r) {
        int m = mbase + r;
        float s0 = bA, s1 = bB;
#pragma unroll
        for (int k = 0; k < FIN; ++k) {
          float xv = xt[m][k];
          s0 = fmaf(xv, wih0_s[cA * FIN + k], s0);
          s1 = fmaf(xv, wih0_s[cB * FIN + k], s1);
        }
        acc0[r] = s0; acc1[r] = s1;
      }
#pragma unroll 2
      for (int kb = 0; kb < 8; ++kb) {
        v16bf a  = load_A_lds(h1buf, lane, kb * 32);
        v16bf w0 = load_B_pk(pk_whh0, ct0 * 8 + kb, lane);
        v16bf w1 = load_B_pk(pk_whh0, ct1 * 8 + kb, lane);
        acc0 = wmma_bf16(a, w0, acc0);
        acc1 = wmma_bf16(a, w1, acc1);
      }
#pragma unroll
      for (int r = 0; r < 8; ++r) {
        gates[(mbase + r) * FH + cA] = acc0[r];
        gates[(mbase + r) * FH + cB] = acc1[r];
      }
    }
    __syncthreads();

    // ---- layer0 cell update ----
#pragma unroll
    for (int r = 0; r < 8; ++r) {
      int j = ej0 + r;
      float ig = sigm(gates[eb * FH + j]);
      float fg = sigm(gates[eb * FH + 256 + j]);
      float gg = tanh_fast(gates[eb * FH + 512 + j]);
      float og = sigm(gates[eb * FH + 768 + j]);
      float c  = fg * c1r[r] + ig * gg;
      c1r[r]   = c;
      h1buf[eb * HSZ + j] = (bf16_t)(og * tanh_fast(c));
    }
    __syncthreads();

    // ---- layer1: gates = h1@Wih1^T + h2@Whh1^T + bias1 ----
#pragma unroll 1
    for (int tp = 0; tp < 2; ++tp) {
      int ct0 = wave * 4 + tp * 2, ct1 = ct0 + 1;
      int cA = ct0 * 16 + colhalf, cB = ct1 * 16 + colhalf;
      v8f acc0, acc1;
      float bA = bias1_s[cA], bB = bias1_s[cB];
#pragma unroll
      for (int r = 0; r < 8; ++r) { acc0[r] = bA; acc1[r] = bB; }
#pragma unroll 2
      for (int kb = 0; kb < 8; ++kb) {
        v16bf a1 = load_A_lds(h1buf, lane, kb * 32);
        v16bf i0 = load_B_pk(pk_wih1, ct0 * 8 + kb, lane);
        v16bf i1 = load_B_pk(pk_wih1, ct1 * 8 + kb, lane);
        acc0 = wmma_bf16(a1, i0, acc0);
        acc1 = wmma_bf16(a1, i1, acc1);
        v16bf a2  = load_A_lds(h2buf, lane, kb * 32);
        v16bf g0  = load_B_pk(pk_whh1, ct0 * 8 + kb, lane);
        v16bf g1  = load_B_pk(pk_whh1, ct1 * 8 + kb, lane);
        acc0 = wmma_bf16(a2, g0, acc0);
        acc1 = wmma_bf16(a2, g1, acc1);
      }
#pragma unroll
      for (int r = 0; r < 8; ++r) {
        gates[(mbase + r) * FH + cA] = acc0[r];
        gates[(mbase + r) * FH + cB] = acc1[r];
      }
    }
    __syncthreads();

    // ---- layer1 cell update ----
#pragma unroll
    for (int r = 0; r < 8; ++r) {
      int j = ej0 + r;
      float ig = sigm(gates[eb * FH + j]);
      float fg = sigm(gates[eb * FH + 256 + j]);
      float gg = tanh_fast(gates[eb * FH + 512 + j]);
      float og = sigm(gates[eb * FH + 768 + j]);
      float c  = fg * c2r[r] + ig * gg;
      c2r[r]   = c;
      h2buf[eb * HSZ + j] = (bf16_t)(og * tanh_fast(c));
    }
    __syncthreads();

    // ---- MLP head: hidden = relu(h2@adv_w1^T + adv_b1), one tile per wave ----
    {
      int ct = wave;                       // 16 waves x 16 tiles over 256 cols
      int cA = ct * 16 + colhalf;
      v8f acc;
      float bA = advb1_s[cA];
#pragma unroll
      for (int r = 0; r < 8; ++r) acc[r] = bA;
#pragma unroll 2
      for (int kb = 0; kb < 8; ++kb) {
        v16bf a  = load_A_lds(h2buf, lane, kb * 32);
        v16bf w0 = load_B_pk(pk_adv1, ct * 8 + kb, lane);
        acc = wmma_bf16(a, w0, acc);
      }
      float* hidden = gates;               // reuse as [BT][HSZ]
#pragma unroll
      for (int r = 0; r < 8; ++r)
        hidden[(mbase + r) * HSZ + cA] = fmaxf(acc[r], 0.0f);
    }
    __syncthreads();

    // ---- score = hidden . adv_w2 + b2 (b2 is softmax-invariant but kept) ----
    {
      const float* hidden = gates;
      float part = 0.0f;
#pragma unroll
      for (int r = 0; r < 8; ++r) {
        int j = ej0 + r;
        part = fmaf(hidden[eb * HSZ + j], advw2_s[j], part);
      }
      sp[tid] = part;                      // sp[b*32 + (tid&31)] == sp[tid]
      __syncthreads();
      if (tid < BT) {
        float s = b2;
#pragma unroll
        for (int q = 0; q < 32; ++q) s += sp[tid * 32 + q];
        scores[(size_t)t * BSZ + row0 + tid] = s;
      }
    }
    __syncthreads();
  }
}

// ---------------------------------------------------------------------------
// Softmax over mixed.reshape(512,1024): rows are contiguous slices of the
// time-major score buffer; output order equals row-major [512,1024].
// ---------------------------------------------------------------------------
__global__ __launch_bounds__(256)
void softmax_rows(const float* __restrict__ scores, float* __restrict__ out) {
  __shared__ float red[256];
  const float* row = scores + (size_t)blockIdx.x * 1024;
  float v[4];
  float vmax = -3.4e38f;
#pragma unroll
  for (int q = 0; q < 4; ++q) { v[q] = row[threadIdx.x + q * 256]; vmax = fmaxf(vmax, v[q]); }
  red[threadIdx.x] = vmax; __syncthreads();
  for (int s = 128; s > 0; s >>= 1) {
    if (threadIdx.x < s) red[threadIdx.x] = fmaxf(red[threadIdx.x], red[threadIdx.x + s]);
    __syncthreads();
  }
  vmax = red[0]; __syncthreads();
  float sum = 0.0f;
#pragma unroll
  for (int q = 0; q < 4; ++q) { v[q] = __expf(v[q] - vmax); sum += v[q]; }
  red[threadIdx.x] = sum; __syncthreads();
  for (int s = 128; s > 0; s >>= 1) {
    if (threadIdx.x < s) red[threadIdx.x] += red[threadIdx.x + s];
    __syncthreads();
  }
  float inv = 1.0f / red[0];
#pragma unroll
  for (int q = 0; q < 4; ++q)
    out[(size_t)blockIdx.x * 1024 + threadIdx.x + q * 256] = v[q] * inv;
}

// ---------------------------------------------------------------------------
extern "C" void kernel_launch(void* const* d_in, const int* in_sizes, int n_in,
                              void* d_out, int out_size, void* d_ws, size_t ws_size,
                              hipStream_t stream) {
  const float* x      = (const float*)d_in[0];
  const float* w_ih0  = (const float*)d_in[1];
  const float* w_hh0  = (const float*)d_in[2];
  const float* b_ih0  = (const float*)d_in[3];
  const float* b_hh0  = (const float*)d_in[4];
  const float* w_ih1  = (const float*)d_in[5];
  const float* w_hh1  = (const float*)d_in[6];
  const float* b_ih1  = (const float*)d_in[7];
  const float* b_hh1  = (const float*)d_in[8];
  const float* adv_w1 = (const float*)d_in[9];
  const float* adv_b1 = (const float*)d_in[10];
  const float* adv_w2 = (const float*)d_in[11];
  const float* adv_b2 = (const float*)d_in[12];

  char* ws = (char*)d_ws;
  bf16_t* pk_whh0 = (bf16_t*)(ws + 0);           // 512KB
  bf16_t* pk_wih1 = (bf16_t*)(ws + (512 << 10)); // 512KB
  bf16_t* pk_whh1 = (bf16_t*)(ws + (1024 << 10));// 512KB
  bf16_t* pk_adv1 = (bf16_t*)(ws + (1536 << 10));// 128KB
  float*  bias0   = (float*)(ws + (1664 << 10)); // 4KB
  float*  bias1   = (float*)(ws + (1668 << 10)); // 4KB
  float*  scores  = (float*)(ws + (1672 << 10)); // 2MB

  pack_wmma_b<<<512, 256, 0, stream>>>(w_hh0,  pk_whh0, HSZ, 8, FH * HSZ);
  pack_wmma_b<<<512, 256, 0, stream>>>(w_ih1,  pk_wih1, HSZ, 8, FH * HSZ);
  pack_wmma_b<<<512, 256, 0, stream>>>(w_hh1,  pk_whh1, HSZ, 8, FH * HSZ);
  pack_wmma_b<<<128, 256, 0, stream>>>(adv_w1, pk_adv1, HSZ, 8, HSZ * HSZ);
  prep_bias<<<4, 256, 0, stream>>>(b_ih0, b_hh0, b_ih1, b_hh1, bias0, bias1);

  lstm_persistent<<<NWG, BLK, 0, stream>>>(x, w_ih0, pk_whh0, pk_wih1, pk_whh1,
                                           pk_adv1, bias0, bias1, adv_b1,
                                           adv_w2, adv_b2, scores);

  softmax_rows<<<512, 256, 0, stream>>>(scores, (float*)d_out);
}
